// NewGRU_1425929142502
// MI455X (gfx1250) — matrix operand order repaired and checked by reading, not compiled, not run on hardware
//
#include <hip/hip_runtime.h>

typedef __bf16 bf16;
typedef __attribute__((ext_vector_type(16))) __bf16 v16bf;
typedef __attribute__((ext_vector_type(8)))  __bf16 v8bf;
typedef __attribute__((ext_vector_type(8)))  float  v8f;

#define BB 64
#define TT 512
#define DD 512
#define HH 512
#define WBLK (512 * 512)   // one 512x512 weight matrix

// ---------------------------------------------------------------------------
// Fragment loaders (wave32; this code's contiguous k = 16*(lane>>4) + s order for A and B).
// A 16x32 bf16 : lane&15 = M, lane>>4 selects K-half -> two 16B runs.
// B 32x16 bf16 : lane&15 = N, one contiguous 32B run.
// ---------------------------------------------------------------------------
static __device__ inline v16bf load_a_bf16(const bf16* __restrict__ ap,
                                           int kb, int half)
{
    v8bf alo = *(const v8bf*)(ap + kb + half * 16);        // K = kb+16h .. +7
    v8bf ahi = *(const v8bf*)(ap + kb + half * 16 + 8);    // K = kb+16h+8 ..
    v16bf a;
#pragma unroll
    for (int j = 0; j < 8; ++j) { a[j] = alo[j]; a[8 + j] = ahi[j]; }
    return a;
}

static __device__ inline v16bf load_a_f32(const float* __restrict__ ap,
                                          int kb, int half)
{
    v8f alo = *(const v8f*)(ap + kb + half * 16);
    v8f ahi = *(const v8f*)(ap + kb + half * 16 + 8);
    v16bf a;
#pragma unroll
    for (int j = 0; j < 8; ++j) { a[j] = (bf16)alo[j]; a[8 + j] = (bf16)ahi[j]; }
    return a;
}

#define WMMA_BF16(A, B, C) \
    __builtin_amdgcn_wmma_f32_16x16x32_bf16(false, (A), false, (B), (short)0, (C), false, false)

// ---------------------------------------------------------------------------
// Single-tile K=512 WMMA dot (A bf16), software-pipelined: loads for ks+1
// issued before the WMMA of ks; SSA double-buffer kills the A WAR hazard.
// ---------------------------------------------------------------------------
static __device__ inline v8f wmma_k512_bf16A(const bf16* __restrict__ A,
                                             const bf16* __restrict__ W,
                                             int mrow, int ncol, int half, v8f acc)
{
    const bf16* ap = A + (size_t)mrow * 512;
    const bf16* bp = W + (size_t)ncol * 512 + half * 16;

    v16bf a_cur = load_a_bf16(ap, 0, half);
    v16bf b_cur = *(const v16bf*)(bp);
#pragma unroll
    for (int ks = 0; ks < 16; ++ks) {
        v16bf a_nxt = a_cur, b_nxt = b_cur;
        if (ks < 15) {
            const int kb2 = (ks + 1) * 32;
            a_nxt = load_a_bf16(ap, kb2, half);
            b_nxt = *(const v16bf*)(bp + kb2);
        }
        acc = WMMA_BF16(a_cur, b_cur, acc);
        a_cur = a_nxt; b_cur = b_nxt;
    }
    return acc;
}

// ---------------------------------------------------------------------------
// 4-tile (16x64) K=512 WMMA dot, f32 A fused-converted to bf16.
// One A fragment feeds 4 independent accumulator chains; A and the 4 B
// fragments are double-buffered across k-steps (software pipeline).
// ---------------------------------------------------------------------------
static __device__ inline void wmma_k512_f32A_x4(const float* __restrict__ A,
                                                const bf16* __restrict__ W,
                                                int mrow, int ncolbase, int l15,
                                                int half, v8f acc[4])
{
    const float* ap = A + (size_t)mrow * 512;
    const bf16* bp0 = W + (size_t)(ncolbase +  0 + l15) * 512 + half * 16;
    const bf16* bp1 = W + (size_t)(ncolbase + 16 + l15) * 512 + half * 16;
    const bf16* bp2 = W + (size_t)(ncolbase + 32 + l15) * 512 + half * 16;
    const bf16* bp3 = W + (size_t)(ncolbase + 48 + l15) * 512 + half * 16;

    v16bf a_cur  = load_a_f32(ap, 0, half);
    v16bf b_cur0 = *(const v16bf*)(bp0);
    v16bf b_cur1 = *(const v16bf*)(bp1);
    v16bf b_cur2 = *(const v16bf*)(bp2);
    v16bf b_cur3 = *(const v16bf*)(bp3);
#pragma unroll
    for (int ks = 0; ks < 16; ++ks) {
        v16bf a_nxt = a_cur;
        v16bf b_nxt0 = b_cur0, b_nxt1 = b_cur1, b_nxt2 = b_cur2, b_nxt3 = b_cur3;
        if (ks < 15) {
            const int kb2 = (ks + 1) * 32;
            a_nxt  = load_a_f32(ap, kb2, half);
            b_nxt0 = *(const v16bf*)(bp0 + kb2);
            b_nxt1 = *(const v16bf*)(bp1 + kb2);
            b_nxt2 = *(const v16bf*)(bp2 + kb2);
            b_nxt3 = *(const v16bf*)(bp3 + kb2);
        }
        acc[0] = WMMA_BF16(a_cur, b_cur0, acc[0]);
        acc[1] = WMMA_BF16(a_cur, b_cur1, acc[1]);
        acc[2] = WMMA_BF16(a_cur, b_cur2, acc[2]);
        acc[3] = WMMA_BF16(a_cur, b_cur3, acc[3]);
        a_cur = a_nxt;
        b_cur0 = b_nxt0; b_cur1 = b_nxt1; b_cur2 = b_nxt2; b_cur3 = b_nxt3;
    }
}

static __device__ inline float fast_sigmoid(float v) {
    return 1.0f / (1.0f + __expf(-v));
}
static __device__ inline float fast_tanh(float v) {
    float a = fabsf(v);
    float e = __expf(-2.0f * a);
    float t = (1.0f - e) / (1.0f + e);
    return copysignf(t, v);
}

// ---------------------------------------------------------------------------
// f32 -> bf16 conversion (weights)
// ---------------------------------------------------------------------------
__global__ void cvt_f32_to_bf16(const float* __restrict__ src,
                                bf16* __restrict__ dst, int n)
{
    int i = blockIdx.x * blockDim.x + threadIdx.x;
    if (i < n) dst[i] = (bf16)src[i];
}

// h_cur = h0 (f32) and bf16 mirror
__global__ void init_hidden(const float* __restrict__ h0,
                            float* __restrict__ h_cur, bf16* __restrict__ h_bf)
{
    int i = blockIdx.x * blockDim.x + threadIdx.x;
    if (i < BB * HH) { float v = h0[i]; h_cur[i] = v; h_bf[i] = (bf16)v; }
}

// ---------------------------------------------------------------------------
// Phase 1: gate_pre[g][t][b][h] =
//   x[b,t,:]·W_ig[h,:] + ctx[b,t,:]·W_pg[h,:] + b_ig[h] + b_pg[h] + b_hg[h]
// Each wave: one 16x64 supertile. Tiles: 3 gates * 2048 Mtiles * 8 Nsupers.
// 8 waves / 256-thread block -> 6144 blocks.
// Wb blocks: 0 W_ir,1 W_hr,2 W_pr,3 W_ii,4 W_hi,5 W_pi,6 W_in,7 W_hn,8 W_pn
// ---------------------------------------------------------------------------
__global__ void gemm_pre(const float* __restrict__ x, const float* __restrict__ ctx,
                         const bf16* __restrict__ Wb,
                         const float* __restrict__ bi_r, const float* __restrict__ bh_r,
                         const float* __restrict__ bp_r,
                         const float* __restrict__ bi_i, const float* __restrict__ bh_i,
                         const float* __restrict__ bp_i,
                         const float* __restrict__ bi_n, const float* __restrict__ bh_n,
                         const float* __restrict__ bp_n,
                         float* __restrict__ gate_pre)
{
    const int wave = threadIdx.x >> 5;
    const int lane = threadIdx.x & 31;
    const int tile = blockIdx.x * (blockDim.x >> 5) + wave;   // 0 .. 49151
    const int g    = tile / (2048 * 8);
    const int rem  = tile % (2048 * 8);
    const int tm   = rem >> 3;         // 0 .. 2047 (16 rows of B*T each)
    const int tn4  = rem & 7;          // supertile: 64 output columns
    const int half = lane >> 4;
    const int l15  = lane & 15;
    const int mrow = tm * 16 + l15;    // row in [B*T, 512]  (b*T + t)
    const int nb   = tn4 * 64;         // first column of supertile

    const bf16* Wi = Wb + (size_t)(3 * g + 0) * WBLK;  // W_i{r,i,n}
    const bf16* Wp = Wb + (size_t)(3 * g + 2) * WBLK;  // W_p{r,i,n}

    v8f acc[4] = {v8f{}, v8f{}, v8f{}, v8f{}};
    wmma_k512_f32A_x4(x,   Wi, mrow, nb, l15, half, acc);
    wmma_k512_f32A_x4(ctx, Wp, mrow, nb, l15, half, acc);

    const float* bi = (g == 0) ? bi_r : (g == 1) ? bi_i : bi_n;
    const float* bh = (g == 0) ? bh_r : (g == 1) ? bh_i : bh_n;
    const float* bp = (g == 0) ? bp_r : (g == 1) ? bp_i : bp_n;

    float* out = gate_pre + (size_t)g * TT * BB * HH;
#pragma unroll
    for (int j = 0; j < 4; ++j) {
        const int col  = nb + j * 16 + l15;
        const float bias = bi[col] + bh[col] + bp[col];
#pragma unroll
        for (int reg = 0; reg < 8; ++reg) {
            int grow = tm * 16 + reg + 8 * half;   // b*T + t
            int bidx = grow >> 9;                  // / T
            int t    = grow & 511;
            out[(size_t)(t * BB + bidx) * HH + col] = acc[j][reg] + bias;
        }
    }
}

// ---------------------------------------------------------------------------
// Phase 2a (per step t): r and i gates.
//   v = h@W_h?^T + gate_pre[?][t]; r-gate -> rh_bf = bf16(sigmoid(v)*h)
//                                  i-gate -> i_gate = sigmoid(v)
// 256 tiles (2 gates * 4 Mtiles * 32 Ntiles), 32 blocks * 8 waves.
// ---------------------------------------------------------------------------
__global__ void step_ri(const bf16* __restrict__ h_bf, const float* __restrict__ h_cur,
                        const bf16* __restrict__ Wb, const float* __restrict__ gate_pre,
                        float* __restrict__ i_gate, bf16* __restrict__ rh_bf, int t)
{
    const int wave = threadIdx.x >> 5;
    const int lane = threadIdx.x & 31;
    const int tile = blockIdx.x * (blockDim.x >> 5) + wave;   // 0 .. 255
    const int g    = tile >> 7;        // 0 = r, 1 = i
    const int rem  = tile & 127;
    const int tm   = rem >> 5;
    const int tn   = rem & 31;
    const int half = lane >> 4;
    const int l15  = lane & 15;
    const int mrow = tm * 16 + l15;
    const int ncol = tn * 16 + l15;

    const bf16* Wh = Wb + (size_t)(3 * g + 1) * WBLK;  // W_hr or W_hi
    v8f acc = {};
    acc = wmma_k512_bf16A(h_bf, Wh, mrow, ncol, half, acc);

    const float* pre = gate_pre + (size_t)g * TT * BB * HH + (size_t)t * BB * HH;
#pragma unroll
    for (int reg = 0; reg < 8; ++reg) {
        int m = tm * 16 + reg + 8 * half;
        size_t idx = (size_t)m * HH + ncol;
        float s = fast_sigmoid(acc[reg] + pre[idx]);
        if (g == 0) {
            rh_bf[idx] = (bf16)(s * h_cur[idx]);
        } else {
            i_gate[idx] = s;
        }
    }
}

// ---------------------------------------------------------------------------
// Phase 2b (per step t): n gate + hidden update + output write.
// 128 tiles, 16 blocks * 8 waves.
// ---------------------------------------------------------------------------
__global__ void step_n(const bf16* __restrict__ rh_bf, const float* __restrict__ i_gate,
                       const bf16* __restrict__ Wb, const float* __restrict__ pre_n,
                       float* __restrict__ h_cur, bf16* __restrict__ h_bf,
                       float* __restrict__ y, float* __restrict__ hT, int t)
{
    const int wave = threadIdx.x >> 5;
    const int lane = threadIdx.x & 31;
    const int tile = blockIdx.x * (blockDim.x >> 5) + wave;   // 0 .. 127
    const int tm   = tile >> 5;
    const int tn   = tile & 31;
    const int half = lane >> 4;
    const int l15  = lane & 15;
    const int mrow = tm * 16 + l15;
    const int ncol = tn * 16 + l15;

    const bf16* Whn = Wb + (size_t)7 * WBLK;
    v8f acc = {};
    acc = wmma_k512_bf16A(rh_bf, Whn, mrow, ncol, half, acc);

    const float* pre = pre_n + (size_t)t * BB * HH;
#pragma unroll
    for (int reg = 0; reg < 8; ++reg) {
        int m = tm * 16 + reg + 8 * half;
        size_t idx = (size_t)m * HH + ncol;
        float n  = fast_tanh(acc[reg] + pre[idx]);
        float ig = i_gate[idx];
        float h  = h_cur[idx];
        float hy = (1.0f - ig) * h + ig * n;
        h_cur[idx] = hy;
        h_bf[idx]  = (bf16)hy;
        y[((size_t)m * TT + t) * HH + ncol] = hy;
        if (t == TT - 1) hT[idx] = hy;
    }
}

// ---------------------------------------------------------------------------
extern "C" void kernel_launch(void* const* d_in, const int* in_sizes, int n_in,
                              void* d_out, int out_size, void* d_ws, size_t ws_size,
                              hipStream_t stream)
{
    (void)in_sizes; (void)n_in; (void)out_size; (void)ws_size;

    const float* x   = (const float*)d_in[0];
    const float* h0  = (const float*)d_in[1];
    const float* ctx = (const float*)d_in[2];
    // weight/bias pairs start at index 3:
    // W_ir b_ir W_hr b_hr W_pr b_pr | W_ii b_ii W_hi b_hi W_pi b_pi | W_in b_in W_hn b_hn W_pn b_pn
    const float* Wf[9];
    const float* bf_[9];
    for (int k = 0; k < 9; ++k) {
        Wf[k]  = (const float*)d_in[3 + 2 * k];
        bf_[k] = (const float*)d_in[4 + 2 * k];
    }
    // order in Wf/bf_: 0 ir, 1 hr, 2 pr, 3 ii, 4 hi, 5 pi, 6 in, 7 hn, 8 pn

    float* y  = (float*)d_out;                       // [B,T,H]
    float* hT = y + (size_t)BB * TT * HH;            // [B,H]

    // ---- workspace carve ----
    char* ws = (char*)d_ws;
    size_t off = 0;
    auto carve = [&](size_t bytes) {
        void* p = ws + off;
        off += (bytes + 255) & ~(size_t)255;
        return p;
    };
    bf16*  Wb       = (bf16*) carve((size_t)9 * WBLK * sizeof(bf16));          // 4.7 MB
    float* gate_pre = (float*)carve((size_t)3 * TT * BB * HH * sizeof(float)); // 201 MB
    float* h_cur    = (float*)carve((size_t)BB * HH * sizeof(float));
    bf16*  h_bf     = (bf16*) carve((size_t)BB * HH * sizeof(bf16));
    bf16*  rh_bf    = (bf16*) carve((size_t)BB * HH * sizeof(bf16));
    float* i_gate   = (float*)carve((size_t)BB * HH * sizeof(float));

    // ---- convert weights to bf16 (L2-resident across all steps) ----
    for (int k = 0; k < 9; ++k)
        cvt_f32_to_bf16<<<WBLK / 256, 256, 0, stream>>>(Wf[k], Wb + (size_t)k * WBLK, WBLK);

    init_hidden<<<(BB * HH) / 256, 256, 0, stream>>>(h0, h_cur, h_bf);

    // ---- phase 1: big WMMA GEMM for the 3 precomputed gate slabs ----
    // 3 gates * 2048 Mtiles * 8 Nsupertiles = 49152 waves, 8 waves/block
    gemm_pre<<<49152 / 8, 256, 0, stream>>>(
        x, ctx, Wb,
        bf_[0], bf_[1], bf_[2],   // b_ir, b_hr, b_pr
        bf_[3], bf_[4], bf_[5],   // b_ii, b_hi, b_pi
        bf_[6], bf_[7], bf_[8],   // b_in, b_hn, b_pn
        gate_pre);

    // ---- phase 2: sequential scan, 2 launches per step (graph-replayed) ----
    const float* pre_n = gate_pre + (size_t)2 * TT * BB * HH;
    for (int t = 0; t < TT; ++t) {
        step_ri<<<32, 256, 0, stream>>>(h_bf, h_cur, Wb, gate_pre, i_gate, rh_bf, t);
        step_n <<<16, 256, 0, stream>>>(rh_bf, i_gate, Wb, pre_n,
                                        h_cur, h_bf, y, hT, t);
    }
}
